// MoleculeGAT_58093727646211
// MI455X (gfx1250) — compile-verified, matrix operand order checked by reading, not run on hardware
//
#include <hip/hip_runtime.h>
#include <hip/hip_bf16.h>

#define N_NODES   20000
#define N_EDGES   320000
#define N_TOTEDGE (N_EDGES + N_NODES)   // with self loops
#define FEAT      64
#define HID       64
#define HEADS     4
#define CH        (HEADS * HID)         // 256
#define N_GRAPHS  512
#define N_TARGETS 3

typedef __attribute__((ext_vector_type(16))) _Float16 v16h;
typedef __attribute__((ext_vector_type(8)))  float    v8f;

// ---------------------------------------------------------------------------
// float <-> monotonic-ordered uint (for atomic max on floats)
// ---------------------------------------------------------------------------
__device__ inline unsigned f2ord(float f) {
    unsigned u = __float_as_uint(f);
    return (u & 0x80000000u) ? ~u : (u | 0x80000000u);
}
__device__ inline float ord2f(unsigned u) {
    return (u & 0x80000000u) ? __uint_as_float(u & 0x7fffffffu)
                             : __uint_as_float(~u);
}

// ---------------------------------------------------------------------------
// zero fill (used for all accumulators each launch; graph-replay safe)
// ---------------------------------------------------------------------------
__global__ void k_zero(float* p, int n) {
    int i = blockIdx.x * blockDim.x + threadIdx.x;
    if (i < n) p[i] = 0.0f;
}

// ---------------------------------------------------------------------------
// WMMA GEMM: C[M x N] = act(A[M x K] @ B[K x N] + bias), all f32 row-major,
// f16 WMMA with f32 accumulate (v_wmma_f32_16x16x32_f16).
//
// Block = 64 threads (2 waves). Block tile = 32 rows x 32 cols of C.
// Each wave owns a 16-row strip and computes two 16x16 tiles (cols 0-15 and
// 16-31 of the panel) from ONE shared A fragment -> 2 WMMA per K-step.
//
// B panel (K x 32) is staged once per block into LDS *in fragment layout*
// (CDNA5 ISA 16-bit B layout: lane = hi*16 + col, halves j -> K = (j&7) +
// hi*8 + (j>=8)*16), so each lane reads its 16 halves as one aligned 32B
// ds_load. K is a template parameter so the K-loop fully unrolls.
//
// Requirements: M % 32 == 0, N % 32 == 0 (20000=625*32, 512=16*32; N=64/256).
// No divergence anywhere -> EXEC all ones for WMMA.
// ---------------------------------------------------------------------------
template <int K, int ACT>
__global__ __launch_bounds__(64)
void k_wmma_gemm(const float* __restrict__ A, const float* __restrict__ B,
                 const float* __restrict__ bias, float* __restrict__ C, int N) {
    constexpr int KSTEPS = K / 32;
    // [kstep][col-subtile][lane][halves] ; each lane row = 32B (b128-aligned)
    __shared__ __align__(32) _Float16 bsh[KSTEPS][2][32][16];

    const int tid  = threadIdx.x;
    const int wave = tid >> 5;
    const int lane = tid & 31;
    const int r    = lane & 15;     // row within A tile / col within B tile
    const int hi   = lane >> 4;     // K-half select
    const int col0 = blockIdx.y * 32;
    const int row0 = blockIdx.x * 32 + wave * 16;

    // ---- stage B panel (K x 32) into fragment-layout LDS (f32 -> f16) ----
    // K*32 elements, 64 threads, K*32 % 64 == 0 -> uniform trip count.
    for (int idx = tid; idx < K * 32; idx += 64) {
        const int k  = idx >> 5;        // 0..K-1
        const int n  = idx & 31;        // 0..31
        const float v = B[(size_t)k * N + col0 + n];
        const int ks  = k >> 5;
        const int kk  = k & 31;
        const int h2  = (kk >> 3) & 1;
        const int j   = (kk & 7) + ((kk & 16) ? 8 : 0);
        bsh[ks][n >> 4][h2 * 16 + (n & 15)][j] = (_Float16)v;
    }
    __syncthreads();

    // ---- K-loop: 1 A fragment + 2 LDS B fragments -> 2 WMMA per step ----
    v8f acc0 = {};
    v8f acc1 = {};
#pragma unroll
    for (int ks = 0; ks < KSTEPS; ++ks) {
        const float* arow = A + (size_t)(row0 + r) * K + ks * 32 + hi * 8;
        v16h af;
#pragma unroll
        for (int j = 0; j < 8; ++j) af[j]     = (_Float16)arow[j];
#pragma unroll
        for (int j = 0; j < 8; ++j) af[8 + j] = (_Float16)arow[16 + j];

        const v16h bf0 = *(const v16h*)&bsh[ks][0][lane][0];
        const v16h bf1 = *(const v16h*)&bsh[ks][1][lane][0];

        acc0 = __builtin_amdgcn_wmma_f32_16x16x32_f16(
            false, af, false, bf0, (short)0, acc0, false, false);
        acc1 = __builtin_amdgcn_wmma_f32_16x16x32_f16(
            false, af, false, bf1, (short)0, acc1, false, false);
    }

    // ---- epilogue: C/D layout: VGPR v -> M = v + hi*8 ; N = lane & 15 ----
    const int cn0 = col0 + r;
    const int cn1 = col0 + 16 + r;
    const float bv0 = bias ? bias[cn0] : 0.0f;
    const float bv1 = bias ? bias[cn1] : 0.0f;
#pragma unroll
    for (int v = 0; v < 8; ++v) {
        const int cm = row0 + hi * 8 + v;
        float v0 = acc0[v] + bv0;
        float v1 = acc1[v] + bv1;
        if (ACT == 1) { v0 = fmaxf(v0, 0.0f); v1 = fmaxf(v1, 0.0f); }
        C[(size_t)cm * N + cn0] = v0;
        C[(size_t)cm * N + cn1] = v1;
    }
}

static void launch_gemm(const float* A, const float* B, const float* bias,
                        float* C, int M, int N, int K, int act,
                        hipStream_t s) {
    dim3 grid(M / 32, N / 32);
    dim3 blk(64);
    if (K == 64) {
        if (act) k_wmma_gemm<64, 1><<<grid, blk, 0, s>>>(A, B, bias, C, N);
        else     k_wmma_gemm<64, 0><<<grid, blk, 0, s>>>(A, B, bias, C, N);
    } else if (K == 256) {
        if (act) k_wmma_gemm<256, 1><<<grid, blk, 0, s>>>(A, B, bias, C, N);
        else     k_wmma_gemm<256, 0><<<grid, blk, 0, s>>>(A, B, bias, C, N);
    } else { // K == 512
        if (act) k_wmma_gemm<512, 1><<<grid, blk, 0, s>>>(A, B, bias, C, N);
        else     k_wmma_gemm<512, 0><<<grid, blk, 0, s>>>(A, B, bias, C, N);
    }
}

// ---------------------------------------------------------------------------
// Per-node attention coefficients: alpha_s[n,h] = <H[n,h,:], a_src[h,:]>
// ---------------------------------------------------------------------------
__global__ void k_alpha(const float* __restrict__ H,
                        const float* __restrict__ a_src,
                        const float* __restrict__ a_dst,
                        float* __restrict__ as_, float* __restrict__ ad_) {
    int idx = blockIdx.x * blockDim.x + threadIdx.x;   // n*HEADS + h
    if (idx >= N_NODES * HEADS) return;
    int n = idx >> 2, h = idx & 3;
    const float* hp = H + (size_t)n * CH + h * HID;
    const float* ws = a_src + h * HID;
    const float* wd = a_dst + h * HID;
    float ss = 0.f, sd = 0.f;
#pragma unroll 8
    for (int f = 0; f < HID; ++f) { float v = hp[f]; ss += v * ws[f]; sd += v * wd[f]; }
    as_[idx] = ss;
    ad_[idx] = sd;
}

// ---------------------------------------------------------------------------
// Edge pass 1: e = leaky_relu(as[src]+ad[dst]); store e; segment max over dst.
// ---------------------------------------------------------------------------
__global__ void k_edge_logits(const int* __restrict__ ei,
                              const float* __restrict__ as_,
                              const float* __restrict__ ad_,
                              float* __restrict__ ebuf,
                              unsigned* __restrict__ mord) {
    int idx = blockIdx.x * blockDim.x + threadIdx.x;   // e*HEADS + h
    if (idx >= N_TOTEDGE * HEADS) return;
    int e = idx >> 2, h = idx & 3;
    int s, d;
    if (e < N_EDGES) { s = ei[e]; d = ei[N_EDGES + e]; }
    else             { s = d = e - N_EDGES; }
    float v = as_[s * HEADS + h] + ad_[d * HEADS + h];
    v = v > 0.f ? v : 0.2f * v;                        // leaky relu
    ebuf[idx] = v;
    atomicMax(&mord[d * HEADS + h], f2ord(v));
}

// ---------------------------------------------------------------------------
// Edge pass 2: ex = exp(e - m[dst]); overwrite ebuf; segment sum denom.
// ---------------------------------------------------------------------------
__global__ void k_edge_exp(const int* __restrict__ ei,
                           const unsigned* __restrict__ mord,
                           float* __restrict__ ebuf,
                           float* __restrict__ denom) {
    int idx = blockIdx.x * blockDim.x + threadIdx.x;
    if (idx >= N_TOTEDGE * HEADS) return;
    int e = idx >> 2, h = idx & 3;
    int d = (e < N_EDGES) ? ei[N_EDGES + e] : (e - N_EDGES);
    float m  = ord2f(mord[d * HEADS + h]);
    float ex = expf(ebuf[idx] - m);
    ebuf[idx] = ex;
    atomicAdd(&denom[d * HEADS + h], ex);
}

// ---------------------------------------------------------------------------
// Edge pass 3: agg[dst,h,f] += H[src,h,f] * ex/(denom[dst,h]+1e-16).
// Thread per (edge, head, 4-float chunk). Working set lives in 192MB L2.
// ---------------------------------------------------------------------------
__global__ void k_edge_aggregate(const int* __restrict__ ei,
                                 const float* __restrict__ H,
                                 const float* __restrict__ ebuf,
                                 const float* __restrict__ denom,
                                 float* __restrict__ agg) {
    int idx = blockIdx.x * blockDim.x + threadIdx.x;   // ((e*4+h)*16)+q
    if (idx >= N_TOTEDGE * HEADS * 16) return;
    int q  = idx & 15;
    int eh = idx >> 4;
    int e  = eh >> 2, h = eh & 3;
    int s, d;
    if (e < N_EDGES) { s = ei[e]; d = ei[N_EDGES + e]; }
    else             { s = d = e - N_EDGES; }
    float alpha = ebuf[eh] / (denom[d * HEADS + h] + 1e-16f);
    const float4 hv = *(const float4*)(H + (size_t)s * CH + h * HID + q * 4);
    float* op = agg + (size_t)d * CH + h * HID + q * 4;
    atomicAdd(op + 0, hv.x * alpha);
    atomicAdd(op + 1, hv.y * alpha);
    atomicAdd(op + 2, hv.z * alpha);
    atomicAdd(op + 3, hv.w * alpha);
}

// ---------------------------------------------------------------------------
// BatchNorm stats: one block per channel, reduce sum / sumsq over nodes.
// (gat_b bias is dropped: a per-column constant cancels exactly in BN.)
// ---------------------------------------------------------------------------
__global__ void k_bn_stats(const float* __restrict__ X,
                           float* __restrict__ mu, float* __restrict__ rsig) {
    const int c = blockIdx.x;
    const int t = threadIdx.x;
    float s = 0.f, s2 = 0.f;
    for (int n = t; n < N_NODES; n += blockDim.x) {
        float v = X[(size_t)n * CH + c];
        s += v; s2 += v * v;
    }
    __shared__ float sh[256], sh2[256];
    sh[t] = s; sh2[t] = s2;
    __syncthreads();
    for (int o = blockDim.x >> 1; o > 0; o >>= 1) {
        if (t < o) { sh[t] += sh[t + o]; sh2[t] += sh2[t + o]; }
        __syncthreads();
    }
    if (t == 0) {
        float m   = sh[0] * (1.0f / N_NODES);
        float var = sh2[0] * (1.0f / N_NODES) - m * m;
        mu[c]   = m;
        rsig[c] = rsqrtf(var + 1e-5f);
    }
}

__global__ void k_bn_elu(const float* __restrict__ X,
                         const float* __restrict__ mu, const float* __restrict__ rsig,
                         const float* __restrict__ g,  const float* __restrict__ b,
                         float* __restrict__ Y) {
    int i = blockIdx.x * blockDim.x + threadIdx.x;
    if (i >= N_NODES * CH) return;
    int c = i & (CH - 1);
    float y = (X[i] - mu[c]) * rsig[c] * g[c] + b[c];
    Y[i] = y > 0.f ? y : (expf(y) - 1.0f);             // ELU
}

// ---------------------------------------------------------------------------
// Graph pooling: mean + max over nodes of each graph.
// ---------------------------------------------------------------------------
__global__ void k_pool_accum(const float* __restrict__ X,
                             const int* __restrict__ batch,
                             float* __restrict__ xsum,
                             unsigned* __restrict__ xmax,
                             float* __restrict__ counts) {
    int idx = blockIdx.x * blockDim.x + threadIdx.x;   // n*64 + q (q = 4-float chunk)
    if (idx >= N_NODES * (CH / 4)) return;
    int n = idx >> 6, q = idx & 63;
    int g = batch[n];
    const float4 v = *(const float4*)(X + (size_t)n * CH + q * 4);
    float* sp = xsum + (size_t)g * CH + q * 4;
    unsigned* mp = xmax + (size_t)g * CH + q * 4;
    atomicAdd(sp + 0, v.x); atomicMax(mp + 0, f2ord(v.x));
    atomicAdd(sp + 1, v.y); atomicMax(mp + 1, f2ord(v.y));
    atomicAdd(sp + 2, v.z); atomicMax(mp + 2, f2ord(v.z));
    atomicAdd(sp + 3, v.w); atomicMax(mp + 3, f2ord(v.w));
    if (q == 0) atomicAdd(&counts[g], 1.0f);
}

__global__ void k_pool_finish(const float* __restrict__ xsum,
                              const unsigned* __restrict__ xmax,
                              const float* __restrict__ counts,
                              float* __restrict__ pooled) {
    int idx = blockIdx.x * blockDim.x + threadIdx.x;   // g*CH + c
    if (idx >= N_GRAPHS * CH) return;
    int g = idx >> 8, c = idx & 255;
    float cnt  = counts[g];
    float mean = xsum[idx] / fmaxf(cnt, 1.0f);
    float mx   = cnt > 0.f ? ord2f(xmax[idx]) : 0.0f;
    pooled[(size_t)g * (2 * CH) + c]      = mean;
    pooled[(size_t)g * (2 * CH) + CH + c] = mx;
}

// ---------------------------------------------------------------------------
// Final tiny layer: out[g,t] = Hh[g,:] @ w2[:,t] + b2[t]
// ---------------------------------------------------------------------------
__global__ void k_head_out(const float* __restrict__ Hh,
                           const float* __restrict__ w2,
                           const float* __restrict__ b2,
                           float* __restrict__ out) {
    int idx = blockIdx.x * blockDim.x + threadIdx.x;
    if (idx >= N_GRAPHS * N_TARGETS) return;
    int g = idx / N_TARGETS, t = idx % N_TARGETS;
    float s = b2[t];
#pragma unroll 8
    for (int f = 0; f < HID; ++f) s += Hh[(size_t)g * HID + f] * w2[f * N_TARGETS + t];
    out[idx] = s;
}

// ---------------------------------------------------------------------------
// Orchestration
// ---------------------------------------------------------------------------
static inline dim3 g1(int n, int b = 256) { return dim3((n + b - 1) / b); }

extern "C" void kernel_launch(void* const* d_in, const int* in_sizes, int n_in,
                              void* d_out, int out_size, void* d_ws, size_t ws_size,
                              hipStream_t stream) {
    const float* x      = (const float*)d_in[0];
    const int*   ei     = (const int*)  d_in[1];
    const int*   batch  = (const int*)  d_in[2];
    const float* w_in   = (const float*)d_in[3];
    const float* b_in   = (const float*)d_in[4];
    const float *gat_w[3], *att_s[3], *att_d[3], *bn_g[3], *bn_b[3];
    for (int l = 0; l < 3; ++l) {
        int base = 5 + l * 6;
        gat_w[l] = (const float*)d_in[base + 0];
        att_s[l] = (const float*)d_in[base + 1];
        att_d[l] = (const float*)d_in[base + 2];
        // base+3 = gat_b: exactly cancelled by batch norm, skipped
        bn_g[l]  = (const float*)d_in[base + 4];
        bn_b[l]  = (const float*)d_in[base + 5];
    }
    const float* head_w1 = (const float*)d_in[23];
    const float* head_b1 = (const float*)d_in[24];
    const float* head_w2 = (const float*)d_in[25];
    const float* head_b2 = (const float*)d_in[26];
    float* out = (float*)d_out;

    // workspace carve (256B aligned chunks)
    char* cur = (char*)d_ws;
    auto carve = [&](size_t bytes) -> void* {
        void* p = cur; cur += (bytes + 255) & ~(size_t)255; return p;
    };
    float*    Xa     = (float*)   carve((size_t)N_NODES * CH * 4);
    float*    Hb     = (float*)   carve((size_t)N_NODES * CH * 4);
    float*    agg    = (float*)   carve((size_t)N_NODES * CH * 4);
    float*    as_    = (float*)   carve((size_t)N_NODES * HEADS * 4);
    float*    ad_    = (float*)   carve((size_t)N_NODES * HEADS * 4);
    float*    ebuf   = (float*)   carve((size_t)N_TOTEDGE * HEADS * 4);
    unsigned* mord   = (unsigned*)carve((size_t)N_NODES * HEADS * 4);
    float*    denom  = (float*)   carve((size_t)N_NODES * HEADS * 4);
    float*    mu     = (float*)   carve(CH * 4);
    float*    rsig   = (float*)   carve(CH * 4);
    float*    counts = (float*)   carve(N_GRAPHS * 4);
    float*    xsum   = (float*)   carve((size_t)N_GRAPHS * CH * 4);
    unsigned* xmax   = (unsigned*)carve((size_t)N_GRAPHS * CH * 4);
    float*    pooled = (float*)   carve((size_t)N_GRAPHS * 2 * CH * 4);
    float*    headh  = (float*)   carve((size_t)N_GRAPHS * HID * 4);

    // 1) input projection + relu : Xa[N x 64] (WMMA, K=64)
    launch_gemm(x, w_in, b_in, Xa, N_NODES, HID, FEAT, 1, stream);

    int din = HID; // layer-0 input width
    for (int l = 0; l < 3; ++l) {
        // 2a) H = X @ W (WMMA, f16 in / f32 acc)
        launch_gemm(Xa, gat_w[l], nullptr, Hb, N_NODES, CH, din, 0, stream);

        // 2b) per-node attention coefficients
        k_alpha<<<g1(N_NODES * HEADS), 256, 0, stream>>>(Hb, att_s[l], att_d[l], as_, ad_);

        // 2c) edge softmax + aggregation (accumulators zeroed every call)
        k_zero<<<g1(N_NODES * HEADS), 256, 0, stream>>>((float*)mord, N_NODES * HEADS);
        k_zero<<<g1(N_NODES * HEADS), 256, 0, stream>>>(denom, N_NODES * HEADS);
        k_zero<<<g1(N_NODES * CH),   256, 0, stream>>>(agg, N_NODES * CH);

        k_edge_logits<<<g1(N_TOTEDGE * HEADS), 256, 0, stream>>>(ei, as_, ad_, ebuf, mord);
        k_edge_exp   <<<g1(N_TOTEDGE * HEADS), 256, 0, stream>>>(ei, mord, ebuf, denom);
        k_edge_aggregate<<<g1(N_TOTEDGE * HEADS * 16), 256, 0, stream>>>(
            ei, Hb, ebuf, denom, agg);

        // 2d) batch norm + ELU -> Xa [N x 256]
        k_bn_stats<<<CH, 256, 0, stream>>>(agg, mu, rsig);
        k_bn_elu<<<g1(N_NODES * CH), 256, 0, stream>>>(agg, mu, rsig, bn_g[l], bn_b[l], Xa);

        din = CH;
    }

    // 3) pooling (mean + max per graph)
    k_zero<<<g1(N_GRAPHS), 256, 0, stream>>>(counts, N_GRAPHS);
    k_zero<<<g1(N_GRAPHS * CH), 256, 0, stream>>>(xsum, N_GRAPHS * CH);
    k_zero<<<g1(N_GRAPHS * CH), 256, 0, stream>>>((float*)xmax, N_GRAPHS * CH);
    k_pool_accum<<<g1(N_NODES * (CH / 4)), 256, 0, stream>>>(Xa, batch, xsum, xmax, counts);
    k_pool_finish<<<g1(N_GRAPHS * CH), 256, 0, stream>>>(xsum, xmax, counts, pooled);

    // 4) head MLP: relu(pooled[512x512] @ w1[512x64] + b1) (WMMA, K=512),
    //    then tiny final layer
    launch_gemm(pooled, head_w1, head_b1, headh, N_GRAPHS, HID, 2 * CH, 1, stream);
    k_head_out<<<g1(N_GRAPHS * N_TARGETS), 256, 0, stream>>>(headh, head_w2, head_b2, out);

    (void)in_sizes; (void)n_in; (void)out_size; (void)ws_size;
}
